// NetMamba_74723841016194
// MI455X (gfx1250) — matrix operand order repaired and checked by reading, not compiled, not run on hardware
//
#include <hip/hip_runtime.h>
#include <hip/hip_bf16.h>
#include <cstdint>

// ---------------- model dims ----------------
constexpr int kB   = 64;
constexpr int kSeq = 401;          // L + 1 (cls token)
constexpr int kL   = 400;
constexpr int kD   = 192;
constexpr int kDI  = 384;
constexpr int kDS  = 16;
constexpr int kDTR = 12;
constexpr int kM   = kB * kSeq;    // 25664 tokens (divisible by 64)
constexpr int kXDBL_LD = 48;       // padded fp32 row stride for x_dbl (44 -> 48)
constexpr int kXDT_LD  = 32;       // padded bf16 row for dt-proj input (12 -> 32)

typedef __attribute__((ext_vector_type(16))) __bf16 v16bf;
typedef __attribute__((ext_vector_type(8)))  float  v8f;

#define USE_ASYNC_LDS 1

// ---------------- helpers ----------------
static __device__ __forceinline__ float siluf(float x) {
    return x / (1.0f + __expf(-x));
}

// 16-element bf16 WMMA fragment from two 16B-aligned LDS chunks
static __device__ __forceinline__ v16bf ld_frag(const __bf16* p0, const __bf16* p1) {
    union { v16bf v; uint4 q[2]; } u;
    u.q[0] = *reinterpret_cast<const uint4*>(p0);
    u.q[1] = *reinterpret_cast<const uint4*>(p1);
    return u.v;
}

// async global -> LDS copy of 16 bytes per lane (ASYNCcnt tracked)
static __device__ __forceinline__ void async_cp16(unsigned lds_byte_off, const void* gaddr) {
#if USE_ASYNC_LDS
    asm volatile("global_load_async_to_lds_b128 %0, %1, off"
                 :: "v"(lds_byte_off), "v"(gaddr) : "memory");
#else
    (void)lds_byte_off; (void)gaddr;
#endif
}
static __device__ __forceinline__ void async_wait0() {
#if USE_ASYNC_LDS
    asm volatile("s_wait_asynccnt 0x0" ::: "memory");
#endif
}

// ---------------- GEMM: C[M,N] = A[M,K] * W[N,Kpad]^T (+epilogue) ----------------
// A bf16 row-major (lda, 16B-aligned rows), W bf16 padded (Npad x Kpad, zero-filled).
// Block = 128 threads (4 waves), tile 64M x 64N, K chunked by 32 (compile-time, unrolled).
// All tile loads are uniform 16B async copies -> no guards anywhere in the hot loop.
enum { EPI_PLAIN = 0, EPI_SPLIT = 1, EPI_SOFTPLUS = 2, EPI_BIAS = 3, EPI_XPROJ = 4 };

constexpr int LSTR = 40;           // LDS row stride in halves (80B; 16B-aligned rows)

template <int EPI, int KPAD>
__global__ __launch_bounds__(128)
void gemm_bf16(const __bf16* __restrict__ A, const __bf16* __restrict__ W,
               float* __restrict__ out0, void* __restrict__ out1v,
               const float* __restrict__ bias,
               int N, int lda, int ldc)
{
    __shared__ __bf16 lA[64 * LSTR];
    __shared__ __bf16 lB[64 * LSTR];

    const int t    = threadIdx.x;
    const int lane = t & 31;
    const int wave = t >> 5;
    const int g    = lane >> 4;
    const int hl   = lane & 15;
    const int bm   = blockIdx.x;
    const int bn   = blockIdx.y;

    v8f acc[4];
    #pragma unroll
    for (int nt = 0; nt < 4; ++nt)
        #pragma unroll
        for (int i = 0; i < 8; ++i) acc[nt][i] = 0.0f;

    // cooperative tile copy: thread -> (row = t>>2 and t>>2 + 32, 16B segment t&3)
    const int crow = t >> 2;                 // 0..31
    const int cseg = t & 3;                  // 0..3 (16B each)
    const unsigned la0 = (unsigned)(size_t)&lA[0] + crow * (LSTR * 2) + cseg * 16;
    const unsigned la1 = la0 + 32 * (LSTR * 2);
    const unsigned lb0 = (unsigned)(size_t)&lB[0] + crow * (LSTR * 2) + cseg * 16;
    const unsigned lb1 = lb0 + 32 * (LSTR * 2);
    const __bf16* gA0 = A + (size_t)(bm * 64 + crow)      * lda  + cseg * 8;
    const __bf16* gA1 = A + (size_t)(bm * 64 + crow + 32) * lda  + cseg * 8;
    const __bf16* gB0 = W + (size_t)(bn * 64 + crow)      * KPAD + cseg * 8;
    const __bf16* gB1 = W + (size_t)(bn * 64 + crow + 32) * KPAD + cseg * 8;

    #pragma unroll
    for (int k0 = 0; k0 < KPAD; k0 += 32) {
#if USE_ASYNC_LDS
        async_cp16(la0, gA0 + k0);
        async_cp16(la1, gA1 + k0);
        async_cp16(lb0, gB0 + k0);
        async_cp16(lb1, gB1 + k0);
        async_wait0();
#else
        *reinterpret_cast<uint4*>(&lA[crow * LSTR + cseg * 8])        = *reinterpret_cast<const uint4*>(gA0 + k0);
        *reinterpret_cast<uint4*>(&lA[(crow + 32) * LSTR + cseg * 8]) = *reinterpret_cast<const uint4*>(gA1 + k0);
        *reinterpret_cast<uint4*>(&lB[crow * LSTR + cseg * 8])        = *reinterpret_cast<const uint4*>(gB0 + k0);
        *reinterpret_cast<uint4*>(&lB[(crow + 32) * LSTR + cseg * 8]) = *reinterpret_cast<const uint4*>(gB1 + k0);
#endif
        __syncthreads();

        // A fragment (16-bit A 16x32: halves 0-7 = K 8g..8g+7, halves 8-15 = K 16+8g..)
        const int arow = wave * 16 + hl;
        v16bf afrag = ld_frag(&lA[arow * LSTR + 8 * g],
                              &lA[arow * LSTR + 16 + 8 * g]);
        #pragma unroll
        for (int nt = 0; nt < 4; ++nt) {
            const int nrow = nt * 16 + hl;
            v16bf bfrag = ld_frag(&lB[nrow * LSTR + 8 * g],
                                  &lB[nrow * LSTR + 16 + 8 * g]);
            acc[nt] = __builtin_amdgcn_wmma_f32_16x16x32_bf16(
                false, afrag, false, bfrag, (short)0, acc[nt], false, false);
        }
        __syncthreads();
    }

    // epilogue: C 16x16 f32: vgpr r, half-wave g -> M = r + 8g, N = lane&15
    const int m0 = bm * 64 + wave * 16 + g * 8;
    #pragma unroll
    for (int nt = 0; nt < 4; ++nt) {
        const int col = bn * 64 + nt * 16 + hl;
        #pragma unroll
        for (int r = 0; r < 8; ++r) {
            const int m = m0 + r;
            float v = acc[nt][r];
            if (EPI == EPI_SOFTPLUS) {
                v += bias[col];
                v = (v > 20.0f) ? v : log1pf(__expf(v));
                out0[(size_t)m * ldc + col] = v;
            } else if (EPI == EPI_BIAS) {
                if (col < N) out0[(size_t)m * ldc + col] = v + bias[col];
            } else if (EPI == EPI_SPLIT) {
                float* z = (float*)out1v;
                if (col < kDI) out0[(size_t)m * ldc + col] = v;
                else           z[(size_t)m * ldc + (col - kDI)] = v;
            } else if (EPI == EPI_XPROJ) {
                __bf16* xdt = (__bf16*)out1v;
                if (col < 44) out0[(size_t)m * ldc + col] = v;
                if (col < kXDT_LD)
                    xdt[(size_t)m * kXDT_LD + col] = (__bf16)((col < kDTR) ? v : 0.0f);
            } else {
                if (col < N) out0[(size_t)m * ldc + col] = v;
            }
        }
    }
}

// ---------------- weight fp32 -> bf16 conversion with zero padding ----------------
__global__ __launch_bounds__(256)
void k_cvt_w(const float* __restrict__ src, __bf16* __restrict__ dst,
             int N, int K, int Kpad, int total)
{
    const int idx = blockIdx.x * 256 + threadIdx.x;
    if (idx >= total) return;
    const int n = idx / Kpad, k = idx % Kpad;
    const float v = (n < N && k < K) ? src[(size_t)n * K + k] : 0.0f;
    dst[idx] = (__bf16)v;
}

// ---------------- patch embed + pos embed + cls token ----------------
__global__ __launch_bounds__(192)
void k_patch(const float* __restrict__ imgs, const float* __restrict__ pw,
             const float* __restrict__ pb, const float* __restrict__ pos,
             const float* __restrict__ clstok, float* __restrict__ resid)
{
    const int token = blockIdx.x;
    const int b = token / kSeq, l = token % kSeq;
    const int d = threadIdx.x;
    float v;
    if (l < kL) {
        const float* ip = imgs + (size_t)b * (kL * 4) + l * 4;
        v = pb[d] + pos[(size_t)l * kD + d]
          + ip[0] * pw[d*4+0] + ip[1] * pw[d*4+1]
          + ip[2] * pw[d*4+2] + ip[3] * pw[d*4+3];
    } else {
        v = clstok[d] + pos[(size_t)kL * kD + d];
    }
    resid[(size_t)token * kD + d] = v;
}

// ---------------- prenorm: resid (+= hidden), normed(bf16) = rmsnorm(resid)*w ----------------
__global__ __launch_bounds__(192)
void k_prenorm(float* __restrict__ resid, const float* __restrict__ hidden,
               const float* __restrict__ w, __bf16* __restrict__ out, int addResid)
{
    __shared__ float red[192];
    const int token = blockIdx.x;
    const int d = threadIdx.x;
    const size_t idx = (size_t)token * kD + d;
    float r = resid[idx];
    if (addResid) { r += hidden[idx]; resid[idx] = r; }
    red[d] = r * r;
    __syncthreads();
    if (d < 96) red[d] += red[d + 96]; __syncthreads();
    if (d < 48) red[d] += red[d + 48]; __syncthreads();
    if (d < 24) red[d] += red[d + 24]; __syncthreads();
    if (d < 12) red[d] += red[d + 12]; __syncthreads();
    if (d < 6)  red[d] += red[d + 6];  __syncthreads();
    if (d < 3)  red[d] += red[d + 3];  __syncthreads();
    const float inv = rsqrtf((red[0] + red[1] + red[2]) * (1.0f / 192.0f) + 1e-5f);
    out[idx] = (__bf16)(r * inv * w[d]);
}

// ---------------- final rmsnorm on cls token only (bf16 out) ----------------
__global__ __launch_bounds__(192)
void k_finalnorm(const float* __restrict__ hidden, const float* __restrict__ resid,
                 const float* __restrict__ w, __bf16* __restrict__ out)
{
    __shared__ float red[192];
    const int b = blockIdx.x;
    const int d = threadIdx.x;
    const size_t idx = ((size_t)b * kSeq + kL) * kD + d;
    const float r = hidden[idx] + resid[idx];
    red[d] = r * r;
    __syncthreads();
    if (d < 96) red[d] += red[d + 96]; __syncthreads();
    if (d < 48) red[d] += red[d + 48]; __syncthreads();
    if (d < 24) red[d] += red[d + 24]; __syncthreads();
    if (d < 12) red[d] += red[d + 12]; __syncthreads();
    if (d < 6)  red[d] += red[d + 6];  __syncthreads();
    if (d < 3)  red[d] += red[d + 3];  __syncthreads();
    const float inv = rsqrtf((red[0] + red[1] + red[2]) * (1.0f / 192.0f) + 1e-5f);
    out[(size_t)b * kD + d] = (__bf16)(r * inv * w[d]);
}

// ---------------- depthwise causal conv(4) + bias + SiLU (bf16 out) ----------------
__global__ __launch_bounds__(256)
void k_conv(const float* __restrict__ xr, const float* __restrict__ cw,
            const float* __restrict__ cb, __bf16* __restrict__ out)
{
    const size_t idx = (size_t)blockIdx.x * 256 + threadIdx.x;
    if (idx >= (size_t)kM * kDI) return;
    const int d = (int)(idx % kDI);
    const size_t token = idx / kDI;
    const int l = (int)(token % kSeq);
    float acc = cb[d];
    #pragma unroll
    for (int j = 0; j < 4; ++j) {
        const int ll = l - 3 + j;
        if (ll >= 0) acc += xr[idx + (size_t)(j - 3) * kDI] * cw[d * 4 + j];
    }
    out[idx] = (__bf16)siluf(acc);
}

// ---------------- fused selective scan + D-skip + SiLU(z) gating ----------------
__global__ __launch_bounds__(256)
void k_scan(const float* __restrict__ alog, const float* __restrict__ dskip,
            const __bf16* __restrict__ xm, const float* __restrict__ z,
            const float* __restrict__ dt, const float* __restrict__ xdbl,
            __bf16* __restrict__ y)
{
    const int tid = blockIdx.x * 256 + threadIdx.x;
    if (tid >= kB * kDI) return;
    const int b = tid / kDI, d = tid % kDI;

    float Arow[kDS], h[kDS];
    #pragma unroll
    for (int s = 0; s < kDS; ++s) {
        Arow[s] = -__expf(alog[(size_t)d * kDS + s]);
        h[s] = 0.0f;
    }
    const float dsk = dskip[d];

    for (int l = 0; l < kSeq; ++l) {
        const size_t base = (size_t)(b * kSeq + l);
        const float dtv = dt[base * kDI + d];
        const float xv  = (float)xm[base * kDI + d];
        const float zv  = z[base * kDI + d];

        union { float4 q[4]; float f[16]; } Bm, Cm;
        const float4* bp = reinterpret_cast<const float4*>(xdbl + base * kXDBL_LD + kDTR);
        const float4* cp = reinterpret_cast<const float4*>(xdbl + base * kXDBL_LD + kDTR + kDS);
        #pragma unroll
        for (int q = 0; q < 4; ++q) { Bm.q[q] = bp[q]; Cm.q[q] = cp[q]; }

        const float dtx = dtv * xv;
        float yv = 0.0f;
        #pragma unroll
        for (int s = 0; s < kDS; ++s) {
            const float dA = __expf(dtv * Arow[s]);
            h[s] = dA * h[s] + dtx * Bm.f[s];
            yv += h[s] * Cm.f[s];
        }
        yv += xv * dsk;
        yv *= siluf(zv);
        y[base * kDI + d] = (__bf16)yv;
    }
}

// ---------------- host orchestration ----------------
extern "C" void kernel_launch(void* const* d_in, const int* in_sizes, int n_in,
                              void* d_out, int out_size, void* d_ws, size_t ws_size,
                              hipStream_t stream)
{
    (void)in_sizes; (void)n_in; (void)out_size; (void)ws_size;

    const float* imgs    = (const float*)d_in[0];
    const float* patch_w = (const float*)d_in[1];
    const float* patch_b = (const float*)d_in[2];
    const float* pos     = (const float*)d_in[3];
    const float* clstok  = (const float*)d_in[4];
    const float* in_w    = (const float*)d_in[5];   // (4,768,192)
    const float* conv_w  = (const float*)d_in[6];   // (4,384,4)
    const float* conv_b  = (const float*)d_in[7];   // (4,384)
    const float* xp_w    = (const float*)d_in[8];   // (4,44,384)
    const float* dt_w    = (const float*)d_in[9];   // (4,384,12)
    const float* dt_b    = (const float*)d_in[10];  // (4,384)
    const float* A_log   = (const float*)d_in[11];  // (4,384,16)
    const float* D_skip  = (const float*)d_in[12];  // (4,384)
    const float* out_w   = (const float*)d_in[13];  // (4,192,384)
    const float* norm_w  = (const float*)d_in[14];  // (4,192)
    const float* normf_w = (const float*)d_in[15];  // (192)
    const float* head_w  = (const float*)d_in[16];  // (1000,192)
    const float* head_b  = (const float*)d_in[17];  // (1000)
    float* out = (float*)d_out;                     // (64,1000)

    // ---- workspace carve-up: fp32 region first (16B aligned), then bf16 region
    float* wsf = (float*)d_ws;
    size_t off = 0;
    auto carvef = [&](size_t n) { float* p = wsf + off; off += n; return p; };
    float* resid  = carvef((size_t)kM * kD);
    float* hidden = carvef((size_t)kM * kD);
    float* xmraw  = carvef((size_t)kM * kDI);   // in_proj xm output (pre-conv)
    float* zbuf   = carvef((size_t)kM * kDI);
    float* xdbl   = carvef((size_t)kM * kXDBL_LD);
    float* dtbuf  = carvef((size_t)kM * kDI);

    __bf16* wsh = (__bf16*)(wsf + off);
    size_t hoff = 0;
    auto carveh = [&](size_t n) { __bf16* p = wsh + hoff; hoff += n; return p; };
    __bf16* normed = carveh((size_t)kM * kD);
    __bf16* xmc    = carveh((size_t)kM * kDI);      // post-conv xm (bf16)
    __bf16* ybuf   = carveh((size_t)kM * kDI);      // gated scan output (bf16)
    __bf16* xdt    = carveh((size_t)kM * kXDT_LD);  // dt-proj input, K padded 12->32
    __bf16* ncls   = carveh((size_t)kB * kD);
    // padded bf16 weights
    __bf16* wbin   = carveh((size_t)4 * 768 * 192);
    __bf16* wbxp   = carveh((size_t)4 * 64 * 384);  // N 44->64
    __bf16* wbdt   = carveh((size_t)4 * 384 * 32);  // K 12->32
    __bf16* wbout  = carveh((size_t)4 * 192 * 384);
    __bf16* wbhead = carveh((size_t)1024 * 192);    // N 1000->1024

    const dim3 blk128(128), blk192(192), blk256(256);

    // 0) convert + pad all weights to bf16 (once per call)
    for (int layer = 0; layer < 4; ++layer) {
        k_cvt_w<<<dim3((768*192 + 255)/256), blk256, 0, stream>>>(
            in_w + (size_t)layer*768*192, wbin + (size_t)layer*768*192, 768, 192, 192, 768*192);
        k_cvt_w<<<dim3((64*384 + 255)/256), blk256, 0, stream>>>(
            xp_w + (size_t)layer*44*384, wbxp + (size_t)layer*64*384, 44, 384, 384, 64*384);
        k_cvt_w<<<dim3((384*32 + 255)/256), blk256, 0, stream>>>(
            dt_w + (size_t)layer*384*12, wbdt + (size_t)layer*384*32, 384, 12, 32, 384*32);
        k_cvt_w<<<dim3((192*384 + 255)/256), blk256, 0, stream>>>(
            out_w + (size_t)layer*192*384, wbout + (size_t)layer*192*384, 192, 384, 384, 192*384);
    }
    k_cvt_w<<<dim3((1024*192 + 255)/256), blk256, 0, stream>>>(
        head_w, wbhead, 1000, 192, 192, 1024*192);

    // 1) patch embed + pos + cls
    k_patch<<<dim3(kM), blk192, 0, stream>>>(imgs, patch_w, patch_b, pos, clstok, resid);

    for (int layer = 0; layer < 4; ++layer) {
        const __bf16* l_inw  = wbin  + (size_t)layer * 768 * 192;
        const __bf16* l_xpw  = wbxp  + (size_t)layer * 64 * 384;
        const __bf16* l_dtw  = wbdt  + (size_t)layer * 384 * 32;
        const __bf16* l_ow   = wbout + (size_t)layer * 192 * 384;
        const float*  l_cw   = conv_w + (size_t)layer * kDI * 4;
        const float*  l_cb   = conv_b + (size_t)layer * kDI;
        const float*  l_dtb  = dt_b   + (size_t)layer * kDI;
        const float*  l_alog = A_log  + (size_t)layer * kDI * kDS;
        const float*  l_dsk  = D_skip + (size_t)layer * kDI;
        const float*  l_nw   = norm_w + (size_t)layer * kD;

        // 2) residual update + rmsnorm (bf16 out)
        k_prenorm<<<dim3(kM), blk192, 0, stream>>>(resid, hidden, l_nw, normed,
                                                   layer == 0 ? 0 : 1);
        // 3) in_proj: (kM,192)bf16 x (768,192)bf16^T -> split xm_raw | z (fp32)
        gemm_bf16<EPI_SPLIT, 192><<<dim3(kM/64, 768/64), blk128, 0, stream>>>(
            normed, l_inw, xmraw, zbuf, nullptr, 2*kDI, kD, kDI);
        // 4) depthwise conv + SiLU -> xmc (bf16)
        k_conv<<<dim3((kM*kDI + 255)/256), blk256, 0, stream>>>(xmraw, l_cw, l_cb, xmc);
        // 5) x_proj: (kM,384) x (64pad,384)^T -> xdbl fp32 (ld 48) + xdt bf16 (K padded)
        gemm_bf16<EPI_XPROJ, 384><<<dim3(kM/64, 1), blk128, 0, stream>>>(
            xmc, l_xpw, xdbl, xdt, nullptr, 44, kDI, kXDBL_LD);
        // 6) dt_proj + softplus: (kM,32pad) x (384,32pad)^T -> dtbuf fp32
        gemm_bf16<EPI_SOFTPLUS, 32><<<dim3(kM/64, kDI/64), blk128, 0, stream>>>(
            xdt, l_dtw, dtbuf, nullptr, l_dtb, kDI, kXDT_LD, kDI);
        // 7) fused selective scan + D-skip + SiLU(z) gate -> ybuf (bf16)
        k_scan<<<dim3((kB*kDI + 255)/256), blk256, 0, stream>>>(
            l_alog, l_dsk, xmc, zbuf, dtbuf, xdbl, ybuf);
        // 8) out_proj: (kM,384) x (192,384)^T -> hidden fp32
        gemm_bf16<EPI_PLAIN, 384><<<dim3(kM/64, kD/64), blk128, 0, stream>>>(
            ybuf, l_ow, hidden, nullptr, nullptr, kD, kDI, kD);
    }

    // 9) final rmsnorm of cls tokens, then head GEMM + bias
    k_finalnorm<<<dim3(kB), blk192, 0, stream>>>(hidden, resid, normf_w, ncls);
    gemm_bf16<EPI_BIAS, 192><<<dim3(kB/64, 1024/64), blk128, 0, stream>>>(
        ncls, wbhead, out, nullptr, head_b, 1000, kD, 1000);
}
